// Cheb_Net_17617955848508
// MI455X (gfx1250) — matrix operand order repaired and checked by reading, not compiled, not run on hardware
//
#include <hip/hip_runtime.h>
#include <hip/hip_bf16.h>

// ---------------------------------------------------------------------------
// ChebNet (K=3) on MI455X / gfx1250.
//  - Sparse props: wave32-per-edge gather + global f32 atomics (L2-resident,
//    192MB L2 holds every 25.6MB feature matrix).
//  - Dense matmuls: v_wmma_f32_16x16x32_f16, B fragments register-resident
//    across a grid-stride row-tile loop. No partial tiles (100000 % 32 == 0),
//    so EXEC is all-ones for every WMMA as the ISA requires.
// ---------------------------------------------------------------------------

typedef __attribute__((ext_vector_type(16))) _Float16 v16h;
typedef __attribute__((ext_vector_type(8)))  float    v8f;

#define N_NODES 100000
#define N_EDGES 1600000
#define F_IN    64
#define HIDDEN  64
#define N_CLASS 32

// ---------------- elementwise / setup kernels ------------------------------

__global__ void fill0_kernel(float* __restrict__ p, int n) {
  int i = blockIdx.x * blockDim.x + threadIdx.x;
  if (i < n) p[i] = 0.0f;
}

__global__ void deg_kernel(const long long* __restrict__ rows,
                           float* __restrict__ deg, int nE) {
  int e = blockIdx.x * blockDim.x + threadIdx.x;
  if (e < nE) atomicAdd(&deg[(int)rows[e]], 1.0f);
}

__global__ void dinv_kernel(float* __restrict__ deg, int n) {
  int i = blockIdx.x * blockDim.x + threadIdx.x;
  if (i < n) {
    float d = deg[i];
    deg[i] = (d > 0.0f) ? rsqrtf(fmaxf(d, 1.0f)) : 0.0f;
  }
}

__global__ void norm_kernel(const long long* __restrict__ rows,
                            const long long* __restrict__ cols,
                            const float* __restrict__ dinv,
                            float* __restrict__ nrm, int nE) {
  int e = blockIdx.x * blockDim.x + threadIdx.x;
  if (e < nE) nrm[e] = -(dinv[(int)rows[e]] * dinv[(int)cols[e]]);
}

// dst[row] += norm[e] * src[col] ; one wave32 per edge, 2 features per lane.
__global__ void prop_kernel(const float* __restrict__ src,
                            float* __restrict__ dst,
                            const long long* __restrict__ rows,
                            const long long* __restrict__ cols,
                            const float* __restrict__ nrm, int nE) {
  int gw   = (blockIdx.x * blockDim.x + threadIdx.x) >> 5;
  int lane = threadIdx.x & 31;
  if (gw < nE) {
    int   r = (int)rows[gw];
    int   c = (int)cols[gw];
    float w = nrm[gw];
    float v0 = src[(size_t)c * 64 + lane];
    float v1 = src[(size_t)c * 64 + 32 + lane];
    atomicAdd(&dst[(size_t)r * 64 + lane],      w * v0);
    atomicAdd(&dst[(size_t)r * 64 + 32 + lane], w * v1);
  }
}

// Chebyshev recursion tail: t2 = 2*t2 - t0
__global__ void cheb2_kernel(float* __restrict__ t2,
                             const float* __restrict__ t0, int n) {
  int i = blockIdx.x * blockDim.x + threadIdx.x;
  if (i < n) t2[i] = 2.0f * t2[i] - t0[i];
}

// ---------------- fused 3-term WMMA matmul ---------------------------------

__device__ __forceinline__ bool keep_mask(unsigned v) {
  v = v * 747796405u + 2891336453u;
  v = ((v >> ((v >> 28u) + 4u)) ^ v) * 277803737u;
  v = (v >> 22u) ^ v;
  return (v & 1u) != 0u;
}

// out[m, n] = sum_t  At[m, :] @ W[t][:, n]  (+bias, optional relu+dropout)
// A matrices are [*, 64] f32 row-major; W is [3][64][NOUT] f32; NOUT in {64,32}.
template <int NOUT, bool RELU_DROP>
__global__ __launch_bounds__(256) void cheb_mm_kernel(
    const float* __restrict__ A0, const float* __restrict__ A1,
    const float* __restrict__ A2, const float* __restrict__ W,
    const float* __restrict__ bias, float* __restrict__ out, int nRowTiles) {
  constexpr int COL_TILES = NOUT / 16;   // 4 (layer1) or 2 (layer2)
  constexpr int RTPB      = 8 / COL_TILES;
  const int w    = threadIdx.x >> 5;
  const int lane = threadIdx.x & 31;
  const int lr   = lane & 15;
  const int hi   = lane >> 4;
  const int ct   = w % COL_TILES;
  const int rtl  = w / COL_TILES;
  const int n0   = ct * 16;

  // B fragments: 3 terms x 2 K-slices, kept in VGPRs for the whole row loop.
  // 16-bit B (32x16) order chosen here: lanes 0-15 hold K=0..15 (2 per VGPR), lanes
  // 16-31 hold K=16..31; column N = lane&15.
  v16h Bf[3][2];
#pragma unroll
  for (int t = 0; t < 3; ++t) {
    const float* Wt = W + t * 64 * NOUT;
#pragma unroll
    for (int s = 0; s < 2; ++s) {
      const int kb = s * 32 + hi * 16;
      v16h b;
#pragma unroll
      for (int e = 0; e < 16; ++e)
        b[e] = (_Float16)Wt[(kb + e) * NOUT + n0 + lr];
      Bf[t][s] = b;
    }
  }
  const float bcol = bias[n0 + lr];

  const float* As[3] = {A0, A1, A2};

  for (int rt = blockIdx.x * RTPB + rtl; rt < nRowTiles;
       rt += gridDim.x * RTPB) {
    const int m0 = rt * 16;
    v8f c = {};
#pragma unroll
    for (int t = 0; t < 3; ++t) {
      const float* Arow = As[t] + (size_t)(m0 + lr) * 64;
#pragma unroll
      for (int s = 0; s < 2; ++s) {
        // 16-bit A (16x32) order chosen here: M = lane&15; lanes0-15 take K = {0..15},
        // lanes16-31 take K = {16..31} (matching B) -> one contiguous
        // 16-float run per lane.
        const float* p0 = Arow + s * 32 + hi * 16;
        const float* p1 = p0 + 8;
        v16h a;
#pragma unroll
        for (int e = 0; e < 8; ++e) {
          a[e]     = (_Float16)p0[e];
          a[e + 8] = (_Float16)p1[e];
        }
        c = __builtin_amdgcn_wmma_f32_16x16x32_f16(
            /*neg_a=*/false, a, /*neg_b=*/false, Bf[t][s],
            /*c_mod=*/(short)0, c, /*reuse_a=*/false, /*reuse_b=*/false);
      }
    }
    // C/D f32 layout: VGPR r -> row m0 + r + 8*hi, col n0 + lr.
    const int n = n0 + lr;
#pragma unroll
    for (int r = 0; r < 8; ++r) {
      const int row = m0 + r + hi * 8;
      float v = c[r] + bcol;
      if (RELU_DROP) {
        v = fmaxf(v, 0.0f);
        v = keep_mask((unsigned)(row * NOUT + n) * 2654435761u) ? v * 2.0f
                                                                : 0.0f;
      }
      out[(size_t)row * NOUT + n] = v;
    }
  }
}

// ---------------- log_softmax: one wave per node (32 lanes = 32 classes) ---

__global__ void lsm_kernel(const float* __restrict__ in,
                           float* __restrict__ out, int nNodes) {
  int gw   = (blockIdx.x * blockDim.x + threadIdx.x) >> 5;
  int lane = threadIdx.x & 31;
  if (gw < nNodes) {
    float v = in[(size_t)gw * N_CLASS + lane];
    float m = v;
#pragma unroll
    for (int o = 16; o > 0; o >>= 1) m = fmaxf(m, __shfl_xor(m, o, 32));
    float e = __expf(v - m);
    float s = e;
#pragma unroll
    for (int o = 16; o > 0; o >>= 1) s += __shfl_xor(s, o, 32);
    out[(size_t)gw * N_CLASS + lane] = v - m - __logf(s);
  }
}

// ---------------------------------------------------------------------------

extern "C" void kernel_launch(void* const* d_in, const int* in_sizes, int n_in,
                              void* d_out, int out_size, void* d_ws,
                              size_t ws_size, hipStream_t stream) {
  const float*     x   = (const float*)d_in[0];
  const long long* ei  = (const long long*)d_in[1];  // int64 [2, N_EDGES]
  const float*     W1  = (const float*)d_in[2];      // [3,64,64]
  const float*     b1  = (const float*)d_in[3];      // [64]
  const float*     W2  = (const float*)d_in[4];      // [3,64,32]
  const float*     b2  = (const float*)d_in[5];      // [32]
  float*           out = (float*)d_out;              // [100000,32]
  float*           ws  = (float*)d_ws;

  const long long* rows = ei;
  const long long* cols = ei + N_EDGES;

  const size_t NF = (size_t)N_NODES * 64;  // 6.4M floats per feature matrix
  float* deg = ws;                      // 100000
  float* nrm = ws + 100352;             // 1.6M
  float* Tx1 = nrm + N_EDGES;           // 6.4M (reused for both layers)
  float* Tx2 = Tx1 + NF;                // 6.4M (reused for both layers)
  float* h   = Tx2 + NF;                // 6.4M
  float* o2  = h + NF;                  // 3.2M

  const int T256 = 256;
  auto blocks = [](size_t n, int t) { return (int)((n + t - 1) / t); };
  const int edgeWaveBlocks = N_EDGES / 8;   // 8 waves (edges) per 256-thr block
  const int nodeWaveBlocks = N_NODES / 8;

  // --- normalization ---
  fill0_kernel<<<blocks(N_NODES, T256), T256, 0, stream>>>(deg, N_NODES);
  deg_kernel<<<blocks(N_EDGES, T256), T256, 0, stream>>>(rows, deg, N_EDGES);
  dinv_kernel<<<blocks(N_NODES, T256), T256, 0, stream>>>(deg, N_NODES);
  norm_kernel<<<blocks(N_EDGES, T256), T256, 0, stream>>>(rows, cols, deg, nrm,
                                                          N_EDGES);

  // --- layer 1: Tx1 = L x ; Tx2 = 2 L Tx1 - x ---
  fill0_kernel<<<blocks(NF, T256), T256, 0, stream>>>(Tx1, (int)NF);
  prop_kernel<<<edgeWaveBlocks, T256, 0, stream>>>(x, Tx1, rows, cols, nrm,
                                                   N_EDGES);
  fill0_kernel<<<blocks(NF, T256), T256, 0, stream>>>(Tx2, (int)NF);
  prop_kernel<<<edgeWaveBlocks, T256, 0, stream>>>(Tx1, Tx2, rows, cols, nrm,
                                                   N_EDGES);
  cheb2_kernel<<<blocks(NF, T256), T256, 0, stream>>>(Tx2, x, (int)NF);

  // h = dropout(relu(x@W1_0 + Tx1@W1_1 + Tx2@W1_2 + b1))
  cheb_mm_kernel<64, true><<<1024, T256, 0, stream>>>(x, Tx1, Tx2, W1, b1, h,
                                                      N_NODES / 16);

  // --- layer 2: Tx1 = L h ; Tx2 = 2 L Tx1 - h ---
  fill0_kernel<<<blocks(NF, T256), T256, 0, stream>>>(Tx1, (int)NF);
  prop_kernel<<<edgeWaveBlocks, T256, 0, stream>>>(h, Tx1, rows, cols, nrm,
                                                   N_EDGES);
  fill0_kernel<<<blocks(NF, T256), T256, 0, stream>>>(Tx2, (int)NF);
  prop_kernel<<<edgeWaveBlocks, T256, 0, stream>>>(Tx1, Tx2, rows, cols, nrm,
                                                   N_EDGES);
  cheb2_kernel<<<blocks(NF, T256), T256, 0, stream>>>(Tx2, h, (int)NF);

  // o2 = h@W2_0 + Tx1@W2_1 + Tx2@W2_2 + b2
  cheb_mm_kernel<32, false><<<1024, T256, 0, stream>>>(h, Tx1, Tx2, W2, b2, o2,
                                                       N_NODES / 16);

  // log_softmax
  lsm_kernel<<<nodeWaveBlocks, T256, 0, stream>>>(o2, out, N_NODES);
}
